// Self_Attn_51161650430034
// MI455X (gfx1250) — matrix operand order repaired and checked by reading, not compiled, not run on hardware
//
#include <hip/hip_runtime.h>

typedef __attribute__((ext_vector_type(16))) _Float16 v16h;
typedef __attribute__((ext_vector_type(8)))  _Float16 v8h;
typedef __attribute__((ext_vector_type(8)))  float    v8f;

#define LDW  72   // ld for 64-wide f16 rows (weights, Mt, M2t, Qt)
#define LDS32 40  // ld for 32-wide f16 rows (Wb1b/Wb2b, T)
#define LDE  132  // f32 energy / out_x scratch
#define LDP  136  // f16 Pn
#define LDV  136  // f32 v2 (c-major)

__device__ __forceinline__ float mishf(float x) {
    float sp = (x > 20.0f) ? x : log1pf(__expf(x));
    return x * tanhf(sp);
}

// A-operand fragment (16x32; this file's k order, matching frag_b): lane l16 = M-row, half h covers
// K = h*16 + 0..15 contiguous. Source row-major [m][k], ld%8==0.
__device__ __forceinline__ v16h frag_a(const _Float16* base, int row, int ld,
                                       int kbase, int l16, int half) {
    const _Float16* p = base + (row + l16) * ld + kbase + half * 16;
    v8h lo = *(const v8h*)p;
    v8h hi = *(const v8h*)(p + 8);
    return __builtin_shufflevector(lo, hi, 0,1,2,3,4,5,6,7,8,9,10,11,12,13,14,15);
}

// B-operand fragment (32x16; this file's k order, matching frag_a): lane l16 = N-col, half h covers
// K = h*16 + 0..15 contiguous. Source row-major [n][k], ld%8==0.
__device__ __forceinline__ v16h frag_b(const _Float16* base, int col, int ld,
                                       int kbase, int l16, int half) {
    const _Float16* p = base + (col + l16) * ld + kbase + half * 16;
    v8h lo = *(const v8h*)p;
    v8h hi = *(const v8h*)(p + 8);
    return __builtin_shufflevector(lo, hi, 0,1,2,3,4,5,6,7,8,9,10,11,12,13,14,15);
}

// B-operand fragment from f32 source with inline cvt to f16
__device__ __forceinline__ v16h frag_b_f32(const float* base, int col, int ld,
                                           int kbase, int l16, int half) {
    const float* p = base + (col + l16) * ld + kbase + half * 16;
    v16h f;
#pragma unroll
    for (int i = 0; i < 16; ++i) f[i] = (_Float16)p[i];
    return f;
}

#define WMMA(acc, a, b) \
    acc = __builtin_amdgcn_wmma_f32_16x16x32_f16(false, (a), false, (b), (short)0, (acc), false, false)

__global__ __launch_bounds__(256)
void self_attn3d_kernel(const float* __restrict__ x,
                        const float* __restrict__ Wq,   const float* __restrict__ Wr1,
                        const float* __restrict__ Wb1a, const float* __restrict__ Wb1b,
                        const float* __restrict__ Wb2a, const float* __restrict__ Wb2b,
                        const float* __restrict__ beta,
                        float* __restrict__ out)
{
    __shared__ __align__(16) _Float16 sWq [64 * LDW];
    __shared__ __align__(16) _Float16 sWr1[64 * LDW];
    __shared__ __align__(16) _Float16 sWb1a[32 * LDW];
    __shared__ __align__(16) _Float16 sWb2a[32 * LDW];
    __shared__ __align__(16) _Float16 sWb1b[64 * LDS32];
    __shared__ __align__(16) _Float16 sWb2b[64 * LDS32];
    __shared__ __align__(16) _Float16 sMt [128 * LDW];   // mish(x), [w][c]
    __shared__ __align__(16) _Float16 sM2t[128 * LDW];   // mish(v1), [w][c]
    __shared__ __align__(16) _Float16 sQt [128 * LDW];   // Q, [w][c]
    __shared__ __align__(16) _Float16 sT  [128 * LDS32]; // T1 then T2, [w][a]
    __shared__ __align__(16) float    sE  [128 * LDE];   // energy f32; reused for out_x
    __shared__ __align__(16) _Float16 sPn [128 * LDP];   // Pn[w][j] = S[j][w]
    __shared__ __align__(16) float    sV2 [64 * LDV];    // v2 f32, [c][w]
    __shared__ __align__(16) float    sMx[128];
    __shared__ __align__(16) float    sRinv[128];

    const int t    = threadIdx.x;
    const int wave = t >> 5;
    const int lane = t & 31;
    const int half = lane >> 4;
    const int l16  = lane & 15;

    // slice (b,d,h): x/out layout [B=2][C=64][D=32][H=128][W=128]
    const int s = blockIdx.x;            // 0..8191
    const int b = s >> 12;
    const int d = (s >> 7) & 31;
    const int h = s & 127;
    const long long base = (long long)b * 33554432LL + (long long)d * 16384LL + (long long)h * 128LL;
    const long long cs   = 524288LL;     // channel stride = D*H*W

    // ---- P0: stage weights as f16 (row-major [c_out][c_in]) ----
    for (int i = t; i < 64 * 64; i += 256) sWq  [(i >> 6) * LDW  + (i & 63)] = (_Float16)Wq[i];
    for (int i = t; i < 64 * 64; i += 256) sWr1 [(i >> 6) * LDW  + (i & 63)] = (_Float16)Wr1[i];
    for (int i = t; i < 32 * 64; i += 256) sWb1a[(i >> 6) * LDW  + (i & 63)] = (_Float16)Wb1a[i];
    for (int i = t; i < 32 * 64; i += 256) sWb2a[(i >> 6) * LDW  + (i & 63)] = (_Float16)Wb2a[i];
    for (int i = t; i < 64 * 32; i += 256) sWb1b[(i >> 5) * LDS32 + (i & 31)] = (_Float16)Wb1b[i];
    for (int i = t; i < 64 * 32; i += 256) sWb2b[(i >> 5) * LDS32 + (i & 31)] = (_Float16)Wb2b[i];

    // ---- P1: Mt[w][c] = f16(mish(x[c][w])) ; coalesced global reads ----
#pragma unroll 4
    for (int it = 0; it < 32; ++it) {
        int idx = t + it * 256;
        int c = idx >> 7, w = idx & 127;
        float xv = x[base + (long long)c * cs + w];
        sMt[w * LDW + c] = (_Float16)mishf(xv);
    }
    __syncthreads();

    // ---- P2: Qt[w][o] = sum_c Mt[w][c] * Wq[o][c]   (32 tiles, 4/wave) ----
#pragma unroll
    for (int i = 0; i < 4; ++i) {
        int tid = wave * 4 + i, wr = (tid >> 2) * 16, cc = (tid & 3) * 16;
        v8f acc = {};
        WMMA(acc, frag_a(sMt, wr, LDW, 0,  l16, half), frag_b(sWq, cc, LDW, 0,  l16, half));
        WMMA(acc, frag_a(sMt, wr, LDW, 32, l16, half), frag_b(sWq, cc, LDW, 32, l16, half));
#pragma unroll
        for (int v = 0; v < 8; ++v) sQt[(wr + half * 8 + v) * LDW + cc + l16] = (_Float16)acc[v];
    }
    // ---- P3: T1t[w][a] = sum_c Mt[w][c] * Wb1a[a][c]  (16 tiles, 2/wave) ----
#pragma unroll
    for (int i = 0; i < 2; ++i) {
        int tid = wave * 2 + i, wr = (tid >> 1) * 16, cc = (tid & 1) * 16;
        v8f acc = {};
        WMMA(acc, frag_a(sMt, wr, LDW, 0,  l16, half), frag_b(sWb1a, cc, LDW, 0,  l16, half));
        WMMA(acc, frag_a(sMt, wr, LDW, 32, l16, half), frag_b(sWb1a, cc, LDW, 32, l16, half));
#pragma unroll
        for (int v = 0; v < 8; ++v) sT[(wr + half * 8 + v) * LDS32 + cc + l16] = (_Float16)acc[v];
    }
    __syncthreads();

    // ---- P4: v1 = mish(Wr1*M + Wb1b*T1); keep v1 in VGPRs; M2t = mish(v1) ----
    float v1r[4][8];
#pragma unroll
    for (int i = 0; i < 4; ++i) {
        int tid = wave * 4 + i, wr = (tid >> 2) * 16, cc = (tid & 3) * 16;
        v8f acc = {};
        WMMA(acc, frag_a(sMt, wr, LDW,  0,  l16, half), frag_b(sWr1,  cc, LDW,  0,  l16, half));
        WMMA(acc, frag_a(sMt, wr, LDW,  32, l16, half), frag_b(sWr1,  cc, LDW,  32, l16, half));
        WMMA(acc, frag_a(sT,  wr, LDS32, 0, l16, half), frag_b(sWb1b, cc, LDS32, 0, l16, half));
#pragma unroll
        for (int v = 0; v < 8; ++v) {
            float v1 = mishf(acc[v]);
            v1r[i][v] = v1;
            sM2t[(wr + half * 8 + v) * LDW + cc + l16] = (_Float16)mishf(v1);
        }
    }
    // ---- P5: E[i][j] = sum_c Qt[i][c]*Qt[j][c]  (64 tiles, 8/wave) ----
#pragma unroll
    for (int i = 0; i < 8; ++i) {
        int tid = wave * 8 + i, ir = (tid >> 3) * 16, jc = (tid & 7) * 16;
        v8f acc = {};
        WMMA(acc, frag_a(sQt, ir, LDW, 0,  l16, half), frag_b(sQt, jc, LDW, 0,  l16, half));
        WMMA(acc, frag_a(sQt, ir, LDW, 32, l16, half), frag_b(sQt, jc, LDW, 32, l16, half));
#pragma unroll
        for (int v = 0; v < 8; ++v) sE[(ir + half * 8 + v) * LDE + jc + l16] = acc[v];
    }
    __syncthreads();

    // ---- P6: per-row softmax stats (E is symmetric; mx[j] also bounds column j) ----
    if (t < 128) {
        float mx = -1e30f;
        for (int j = 0; j < 128; ++j) mx = fmaxf(mx, sE[t * LDE + j]);
        float ssum = 0.0f;
        for (int j = 0; j < 128; ++j) ssum += __expf(sE[t * LDE + j] - mx);
        sMx[t]   = mx;
        sRinv[t] = 1.0f / ssum;
    }
    __syncthreads();

    // ---- P7: Pn[w][j] = S[j][w] = exp(E[w][j]-mx[j]) * rinv[j]  (symmetry) ----
#pragma unroll 4
    for (int it = 0; it < 64; ++it) {
        int idx = t + it * 256;
        int i = idx >> 7, j = idx & 127;
        sPn[i * LDP + j] = (_Float16)(__expf(sE[i * LDE + j] - sMx[j]) * sRinv[j]);
    }
    __syncthreads();

    // ---- P8: T2t[w][a] = sum_c M2t[w][c] * Wb2a[a][c]  (reuse sT) ----
#pragma unroll
    for (int i = 0; i < 2; ++i) {
        int tid = wave * 2 + i, wr = (tid >> 1) * 16, cc = (tid & 1) * 16;
        v8f acc = {};
        WMMA(acc, frag_a(sM2t, wr, LDW, 0,  l16, half), frag_b(sWb2a, cc, LDW, 0,  l16, half));
        WMMA(acc, frag_a(sM2t, wr, LDW, 32, l16, half), frag_b(sWb2a, cc, LDW, 32, l16, half));
#pragma unroll
        for (int v = 0; v < 8; ++v) sT[(wr + half * 8 + v) * LDS32 + cc + l16] = (_Float16)acc[v];
    }
    __syncthreads();

    // ---- P9: v2 = mish(v1 + Wb2b*T2) -> sV2 [c][w] f32 (C operand = v1 regs) ----
#pragma unroll
    for (int i = 0; i < 4; ++i) {
        int tid = wave * 4 + i, wr = (tid >> 2) * 16, cc = (tid & 3) * 16;
        v8f acc;
#pragma unroll
        for (int v = 0; v < 8; ++v) acc[v] = v1r[i][v];
        WMMA(acc, frag_a(sT, wr, LDS32, 0, l16, half), frag_b(sWb2b, cc, LDS32, 0, l16, half));
#pragma unroll
        for (int v = 0; v < 8; ++v)
            sV2[(cc + l16) * LDV + wr + half * 8 + v] = mishf(acc[v]);
    }
    __syncthreads();

    // ---- P10: Ot[w][c] = sum_j Pn[w][j] * v2[c][j] -> reuse sE as f32 scratch ----
#pragma unroll
    for (int i = 0; i < 4; ++i) {
        int tid = wave * 4 + i, wr = (tid >> 2) * 16, cc = (tid & 3) * 16;
        v8f acc = {};
#pragma unroll
        for (int ks = 0; ks < 4; ++ks) {
            WMMA(acc, frag_a(sPn, wr, LDP, ks * 32, l16, half),
                      frag_b_f32(sV2, cc, LDV, ks * 32, l16, half));
        }
#pragma unroll
        for (int v = 0; v < 8; ++v) sE[(wr + half * 8 + v) * LDE + cc + l16] = acc[v];
    }
    __syncthreads();

    // ---- P11: out[c][w] = beta * Ot[w][c] + v2[c][w] ; coalesced stores ----
    const float bta = beta[0];
#pragma unroll 4
    for (int it = 0; it < 32; ++it) {
        int idx = t + it * 256;
        int c = idx >> 7, w = idx & 127;
        out[base + (long long)c * cs + w] = bta * sE[w * LDE + c] + sV2[c * LDV + w];
    }
}

extern "C" void kernel_launch(void* const* d_in, const int* in_sizes, int n_in,
                              void* d_out, int out_size, void* d_ws, size_t ws_size,
                              hipStream_t stream) {
    (void)in_sizes; (void)n_in; (void)out_size; (void)d_ws; (void)ws_size;
    const float* x    = (const float*)d_in[0];
    const float* Wq   = (const float*)d_in[1];
    const float* Wr1  = (const float*)d_in[2];
    const float* Wb1a = (const float*)d_in[3];
    const float* Wb1b = (const float*)d_in[4];
    const float* Wb2a = (const float*)d_in[5];
    const float* Wb2b = (const float*)d_in[6];
    const float* beta = (const float*)d_in[7];
    float* out = (float*)d_out;

    dim3 grid(2 * 32 * 128);   // one workgroup per (b,d,h) slice
    dim3 block(256);           // 8 wave32 waves
    self_attn3d_kernel<<<grid, block, 0, stream>>>(x, Wq, Wr1, Wb1a, Wb1b,
                                                   Wb2a, Wb2b, beta, out);
}
